// EdgeGNN_15187004358829
// MI455X (gfx1250) — matrix operand order repaired and checked by reading, not compiled, not run on hardware
//
#include <hip/hip_runtime.h>
#include <math.h>

// ---------------------------------------------------------------------------
// EdgeGNN forward for MI455X (gfx1250, wave32).
//
// Bandwidth-bound on [4,2048,2048] edge tensors. Strategy:
//  - fuse leaky-relu * edge_attr with row-sum reduction (1 pass),
//  - DSN identity: lamb^2 - lamb(a+b) + ab = (lamb-a)(lamb-b); denom == lamb,
//  - attention stored bf16 -> WMMA A operand directly (halves GEMM traffic),
//  - att @ Wh via v_wmma_f32_16x16x32_bf16, Wh pre-swizzled to B-lane layout,
//  - A-panels DMA'd to LDS via tensor_load_to_lds (TDM) with D# padding
//    (1 dword per 256 dwords) so LDS row stride = 1028 dwords == 4 mod 64
//    banks -> conflict-free ds_load_b128 fragment gathers,
//  - B operand software-pipelined one K-step ahead in registers so
//    s_wait overlaps WMMA issue.
// ---------------------------------------------------------------------------

constexpr int    kN   = 2048;
constexpr int    kE   = 4;
constexpr int    kD0  = 128;
constexpr int    kD1  = 64;
constexpr int    kD2  = 16;
constexpr int    kNG  = 8;
constexpr size_t kNN  = (size_t)kN * kN;

typedef __attribute__((ext_vector_type(16))) __bf16 v16bf;
typedef __attribute__((ext_vector_type(8)))  __bf16 v8bf;
typedef __attribute__((ext_vector_type(8)))  float  v8f;
typedef unsigned int u32x4 __attribute__((ext_vector_type(4)));
typedef int          i32x4 __attribute__((ext_vector_type(4)));
typedef int          i32x8 __attribute__((ext_vector_type(8)));

#define TDM_OK __has_builtin(__builtin_amdgcn_tensor_load_to_lds)

// LDS panel: 16 rows x 2048 bf16 with 4B pad per 1024B -> 65792 bytes
constexpr unsigned kPanelBytes = 16 * kN * 2 + (16 * kN * 2 / 1024) * 4;

// workspace layout (bytes, all offsets are >=256B aligned)
constexpr size_t OFF_EDGE  = 0;                                   // 64MB fp32 [E][N][N]
constexpr size_t OFF_ATT   = OFF_EDGE  + kE * kNN * 4;            // 32MB bf16 [E][N][N]
constexpr size_t OFF_XCAT  = OFF_ATT   + kE * kNN * 2;            // 8MB  fp32 [N][1024]
constexpr size_t OFF_WH    = OFF_XCAT  + (size_t)kN * 1024 * 4;   // [N][64] fp32
constexpr size_t OFF_BSWZ  = OFF_WH    + (size_t)kN * kD1 * 4;    // swizzled B, bf16
constexpr size_t OFF_WHOUT = OFF_BSWZ  + (size_t)kN * kD1 * 2;    // [N][16] fp32
constexpr size_t OFF_BSWZO = OFF_WHOUT + (size_t)kN * kD2 * 4;    // swizzled B out, bf16
constexpr size_t OFF_S1    = OFF_BSWZO + (size_t)kN * kD2 * 2;
constexpr size_t OFF_S2    = OFF_S1    + kN * 4;
constexpr size_t OFF_ASUM  = OFF_S2    + kN * 4;                  // [E][N]
constexpr size_t OFF_BSUM  = OFF_ASUM  + kE * kN * 4;             // [E][N]
constexpr size_t OFF_CHANS = OFF_BSUM  + kE * kN * 4;             // [E][4] scalars
constexpr size_t OFF_HP    = OFF_CHANS + 256;                     // [N][16]
constexpr size_t OFF_BN    = OFF_HP    + (size_t)kN * kD2 * 4;    // 32 floats
constexpr size_t OFF_XP    = OFF_BN    + 256;                     // [N][16]
constexpr size_t OFF_GS    = OFF_XP    + (size_t)kN * kD2 * 4;    // [8][32]

__device__ __forceinline__ float d_elu(float x) { return x > 0.f ? x : (expf(x) - 1.f); }

// ---- TDM: DMA a [height x width] bf16 2D tile (row stride = stride elems)
// from global into LDS at byte offset lds_off, with 1-dword pad per 256 dwords.
__device__ __forceinline__ void tdm_load_2d_bf16(const void* gsrc, unsigned lds_off,
                                                 unsigned width, unsigned height,
                                                 unsigned stride) {
#if TDM_OK
  const unsigned long long ga = (unsigned long long)(size_t)gsrc;
  u32x4 g0;
  g0.x = 1u;                                                  // count=1 (valid user D#)
  g0.y = lds_off;                                             // lds_addr (bytes)
  g0.z = (unsigned)ga;                                        // global_addr[31:0]
  g0.w = (unsigned)((ga >> 32) & 0x01ffffffu) | (2u << 30);   // addr[56:32] | type=2
  i32x8 g1;
  g1[0] = (int)((1u << 16) |            // data_size = 1 (2 bytes)
                (1u << 20) |            // pad_enable
                (7u << 22));            // pad_interval = 256 dwords (pad_amount=0 -> 1 dw)
  g1[1] = (int)(width << 16);                                 // tensor_dim0[15:0]
  g1[2] = (int)((width >> 16) | (height << 16));              // dim0[31:16] | dim1[15:0]
  g1[3] = (int)((height >> 16) | (width << 16));              // dim1[31:16] | tile_dim0
  g1[4] = (int)height;                                        // tile_dim1 (tile_dim2=0)
  g1[5] = (int)stride;                                        // tensor_dim0_stride[31:0]
  g1[6] = 0;
  g1[7] = 0;
  const i32x4 gz = {0, 0, 0, 0};
#if defined(__clang_major__) && (__clang_major__ >= 23)
  const i32x8 gz8 = {0, 0, 0, 0, 0, 0, 0, 0};
  __builtin_amdgcn_tensor_load_to_lds(g0, g1, gz, gz, gz8, 0);
#else
  __builtin_amdgcn_tensor_load_to_lds(g0, g1, gz, gz, 0);
#endif
#else
  (void)gsrc; (void)lds_off; (void)width; (void)height; (void)stride;
#endif
}

// padded LDS byte offset for raw stream offset (pad: +4B per 1024B)
__device__ __forceinline__ unsigned lds_pad(unsigned raw) {
  return raw + ((raw >> 10) << 2);
}

// -------------------- zero graph accumulator --------------------
__global__ void k_zero_gs(float* gs) { gs[threadIdx.x] = 0.f; }

// -------------------- initial transpose + row sums --------------------
__global__ void k_init_edges(const float* __restrict__ ef, float* __restrict__ dst,
                             float* __restrict__ a_sum) {
  const int n = blockIdx.x, t = threadIdx.x;
  float rp[kE] = {0.f, 0.f, 0.f, 0.f};
  for (int m = t; m < kN; m += 256) {
    const float4 v = ((const float4*)ef)[(size_t)n * kN + m];
    const size_t q = (size_t)n * kN + m;
    dst[q + 0 * kNN] = v.x;  dst[q + 1 * kNN] = v.y;
    dst[q + 2 * kNN] = v.z;  dst[q + 3 * kNN] = v.w;
    rp[0] += v.x; rp[1] += v.y; rp[2] += v.z; rp[3] += v.w;
  }
  __shared__ float red[kE][256];
#pragma unroll
  for (int c = 0; c < kE; ++c) red[c][t] = rp[c];
  __syncthreads();
  for (int s = 128; s > 0; s >>= 1) {
    if (t < s) {
#pragma unroll
      for (int c = 0; c < kE; ++c) red[c][t] += red[c][t + s];
    }
    __syncthreads();
  }
  if (t == 0) {
#pragma unroll
    for (int c = 0; c < kE; ++c) a_sum[c * kN + n] = red[c][0];
  }
}

// -------------------- GAT edge pass: e = lrelu(s1+s2) * edge_attr, + row sums
__global__ void k_edge_gat(const float* __restrict__ src, float* __restrict__ dst,
                           const float* __restrict__ s1, const float* __restrict__ s2,
                           float* __restrict__ a_sum) {
  const int n = blockIdx.x, t = threadIdx.x;
  const float s1n = s1[n];
  float rp[kE] = {0.f, 0.f, 0.f, 0.f};
  for (int m = t; m < kN; m += 256) {
    float lr = s1n + s2[m];
    lr = lr > 0.f ? lr : 0.2f * lr;
    const size_t q = (size_t)n * kN + m;
#pragma unroll
    for (int c = 0; c < kE; ++c) {
      const float v = lr * src[q + (size_t)c * kNN];
      dst[q + (size_t)c * kNN] = v;
      rp[c] += v;
    }
  }
  __shared__ float red[kE][256];
#pragma unroll
  for (int c = 0; c < kE; ++c) red[c][t] = rp[c];
  __syncthreads();
  for (int s = 128; s > 0; s >>= 1) {
    if (t < s) {
#pragma unroll
      for (int c = 0; c < kE; ++c) red[c][t] += red[c][t + s];
    }
    __syncthreads();
  }
  if (t == 0) {
#pragma unroll
    for (int c = 0; c < kE; ++c) a_sum[c * kN + n] = red[c][0];
  }
}

// -------------------- column sums --------------------
__global__ void k_colsum(const float* __restrict__ e, float* __restrict__ b_sum) {
  const int idx = blockIdx.x * 256 + threadIdx.x;  // [0, E*N)
  const int c = idx >> 11, m = idx & (kN - 1);
  const float* p = e + (size_t)c * kNN + m;
  float s = 0.f;
  for (int n = 0; n < kN; ++n) {
    __builtin_prefetch(p + (size_t)(n + 32) * kN, 0, 0);  // global_prefetch_b8
    s += p[(size_t)n * kN];
  }
  b_sum[idx] = s;
}

// -------------------- per-channel DSN scalars: {lamb, 1/r, 1/lamb, S} ------
__global__ void k_scalars(const float* __restrict__ a_sum, const float* __restrict__ b_sum,
                          float* __restrict__ chans) {
  const int c = blockIdx.x, t = threadIdx.x;
  float amax = -3.4e38f, bmax = -3.4e38f, s = 0.f;
  for (int n = t; n < kN; n += 256) {
    const float av = a_sum[c * kN + n];
    amax = fmaxf(amax, av);
    s += av;
    bmax = fmaxf(bmax, b_sum[c * kN + n]);
  }
  __shared__ float sm[3][256];
  sm[0][t] = amax; sm[1][t] = bmax; sm[2][t] = s;
  __syncthreads();
  for (int st = 128; st > 0; st >>= 1) {
    if (t < st) {
      sm[0][t] = fmaxf(sm[0][t], sm[0][t + st]);
      sm[1][t] = fmaxf(sm[1][t], sm[1][t + st]);
      sm[2][t] += sm[2][t + st];
    }
    __syncthreads();
  }
  if (t == 0) {
    const float lamb = fmaxf(sm[0][0], sm[1][0]);
    const float S = sm[2][0];
    const float r = (float)kN * lamb - S;
    chans[c * 4 + 0] = lamb;
    chans[c * 4 + 1] = (r != 0.f) ? 1.f / r : 0.f;
    chans[c * 4 + 2] = (lamb != 0.f) ? 1.f / lamb : 0.f;
    chans[c * 4 + 3] = S;
  }
}

// -------------------- DSN apply, fp32 in-place (initial stage) --------------------
__global__ void k_apply_init(float* __restrict__ e, const float* __restrict__ a_sum,
                             const float* __restrict__ b_sum, const float* __restrict__ chans) {
  const size_t i = (size_t)blockIdx.x * 256 + threadIdx.x;
  const int c = (int)(i / kNN);
  const size_t rem = i - (size_t)c * kNN;
  const int n = (int)(rem >> 11), m = (int)(rem & (kN - 1));
  const float t = e[i];
  const float lamb = chans[c * 4 + 0], invr = chans[c * 4 + 1], invl = chans[c * 4 + 2];
  const float tt = (t + (lamb - a_sum[c * kN + n]) * (lamb - b_sum[c * kN + m]) * invr) * invl;
  e[i] = (t > 0.f) ? tt : t;
}

// -------------------- DSN apply -> bf16 attention --------------------
__global__ void k_apply_bf16(const float* __restrict__ e, __bf16* __restrict__ att,
                             const float* __restrict__ a_sum, const float* __restrict__ b_sum,
                             const float* __restrict__ chans) {
  const size_t i = (size_t)blockIdx.x * 256 + threadIdx.x;
  const int c = (int)(i / kNN);
  const size_t rem = i - (size_t)c * kNN;
  const int n = (int)(rem >> 11), m = (int)(rem & (kN - 1));
  const float t = e[i];
  const float lamb = chans[c * 4 + 0], invr = chans[c * 4 + 1], invl = chans[c * 4 + 2];
  const float tt = (t + (lamb - a_sum[c * kN + n]) * (lamb - b_sum[c * kN + m]) * invr) * invl;
  att[i] = (__bf16)((t > 0.f) ? tt : t);
}

// -------------------- Wh = x @ W  (2048x128 @ 128x64, fp32, tiny) ----------
__global__ void k_gemm_Wh(const float* __restrict__ x, const float* __restrict__ W,
                          float* __restrict__ Wh) {
  const int t = threadIdx.x;
  const int r = blockIdx.x * 4 + (t >> 6);
  const int f = t & 63;
  float acc = 0.f;
#pragma unroll 8
  for (int k = 0; k < kD0; ++k) acc += x[(size_t)r * kD0 + k] * W[(size_t)k * kD1 + f];
  Wh[(size_t)r * kD1 + f] = acc;
}

// -------------------- Wh_out = xcat @ W_out (2048x1024 @ 1024x16) ----------
__global__ void k_gemm_Whout(const float* __restrict__ x, const float* __restrict__ W,
                             float* __restrict__ Wh) {
  const int t = threadIdx.x;
  const int r = blockIdx.x * 16 + (t >> 4);
  const int f = t & 15;
  float acc = 0.f;
#pragma unroll 8
  for (int k = 0; k < 1024; ++k) acc += x[(size_t)r * 1024 + k] * W[(size_t)k * kD2 + f];
  Wh[(size_t)r * kD2 + f] = acc;
}

// -------------------- swizzle Wh into WMMA B-operand lane layout -----------
__global__ void k_swizzleB(const float* __restrict__ Wh, __bf16* __restrict__ dst,
                           int Fo, int ntiles) {
  const int idx = blockIdx.x * 256 + threadIdx.x;  // < 2048*Fo
  const int j = idx & 15;
  const int lane = (idx >> 4) & 31;
  const int rest = idx >> 9;
  const int nt = rest % ntiles;
  const int kt = rest / ntiles;
  const int kpos = ((lane >> 4) << 3) + (j & 7) + ((j >> 3) << 4);
  const int col = nt * 16 + (lane & 15);
  dst[idx] = (__bf16)Wh[(size_t)(kt * 32 + kpos) * Fo + col];
}

// -------------------- s1/s2 = Wh @ a[:Fo], Wh @ a[Fo:] --------------------
__global__ void k_s12(const float* __restrict__ Wh, const float* __restrict__ avec,
                      float* __restrict__ s1, float* __restrict__ s2, int Fo) {
  const int r = blockIdx.x * 256 + threadIdx.x;
  float a1 = 0.f, a2 = 0.f;
  for (int f = 0; f < Fo; ++f) {
    const float w = Wh[(size_t)r * Fo + f];
    a1 += w * avec[f];
    a2 += w * avec[Fo + f];
  }
  s1[r] = a1;
  s2[r] = a2;
}

// -------------------- WMMA GEMM: xcat[:, head] = elu(att_c @ Wh) -----------
// block = 1 wave; grid = (128 row tiles, 4 channels).
// A panel (16x2048 bf16) DMA'd to LDS by the Tensor Data Mover; B operand
// register-pipelined one K-step ahead.
__global__ __launch_bounds__(32)
void k_gemm_att(const __bf16* __restrict__ att, const __bf16* __restrict__ bswz,
                float* __restrict__ xcat, int head) {
  const int rt = blockIdx.x;
  const int c = blockIdx.y;
  const int lane = threadIdx.x;
  const int rloc = lane & 15;
  const int khalf = lane >> 4;

#if TDM_OK
  __shared__ __align__(128) char apanel[kPanelBytes];
  tdm_load_2d_bf16(att + ((size_t)c * kN + (size_t)rt * 16) * kN,
                   (unsigned)(size_t)(void*)apanel, kN, 16, kN);
  asm volatile("s_wait_tensorcnt 0x0" ::: "memory");
  const unsigned rawbase = (unsigned)(rloc * kN + khalf * 8) * 2u;
#else
  const __bf16* abase = att + ((size_t)c * kN + (size_t)rt * 16 + rloc) * kN + khalf * 8;
#endif

  const __bf16* bptr = bswz + ((size_t)lane << 4);
  v16bf bcur[4];
#pragma unroll
  for (int nt = 0; nt < 4; ++nt)
    bcur[nt] = *(const v16bf*)(bptr + (size_t)nt * 512);

  v8f acc[4] = {};
  for (int kt = 0; kt < 64; ++kt) {
#if TDM_OK
    const unsigned raw = rawbase + (unsigned)kt * 64u;
    const v8bf lo = *(const v8bf*)(apanel + lds_pad(raw));
    const v8bf hi = *(const v8bf*)(apanel + lds_pad(raw + 32u));
#else
    const __bf16* ap = abase + kt * 32;
    const v8bf lo = *(const v8bf*)(ap);
    const v8bf hi = *(const v8bf*)(ap + 16);
#endif
    const v16bf a = __builtin_shufflevector(lo, hi, 0, 1, 2, 3, 4, 5, 6, 7,
                                            8, 9, 10, 11, 12, 13, 14, 15);
    const int ktn = (kt + 1) & 63;
    v16bf bnext[4];
#pragma unroll
    for (int nt = 0; nt < 4; ++nt)
      bnext[nt] = *(const v16bf*)(bptr + (size_t)(ktn * 4 + nt) * 512);
#pragma unroll
    for (int nt = 0; nt < 4; ++nt)
      acc[nt] = __builtin_amdgcn_wmma_f32_16x16x32_bf16(false, a, false, bcur[nt],
                                                        (short)0, acc[nt], false, false);
#pragma unroll
    for (int nt = 0; nt < 4; ++nt) bcur[nt] = bnext[nt];
  }

  float* out = xcat + (size_t)(rt * 16) * 1024 + head * 256 + c * 64;
#pragma unroll
  for (int nt = 0; nt < 4; ++nt) {
#pragma unroll
    for (int v = 0; v < 8; ++v) {
      const int r = v + khalf * 8;
      out[(size_t)r * 1024 + nt * 16 + rloc] = d_elu(acc[nt][v]);
    }
  }
}

// -------------------- WMMA GEMM: hp = sum_c att_c @ Wh_out (N=16) ----------
__global__ __launch_bounds__(32)
void k_gemm_out(const __bf16* __restrict__ att, const __bf16* __restrict__ bswz,
                float* __restrict__ hp) {
  const int rt = blockIdx.x;
  const int lane = threadIdx.x;
  const int rloc = lane & 15;
  const int khalf = lane >> 4;
#if TDM_OK
  __shared__ __align__(128) char apanel[kPanelBytes];
  const unsigned rawbase = (unsigned)(rloc * kN + khalf * 8) * 2u;
#endif
  const __bf16* bptr = bswz + ((size_t)lane << 4);
  v8f acc = {};
  for (int c = 0; c < kE; ++c) {
#if TDM_OK
    tdm_load_2d_bf16(att + ((size_t)c * kN + (size_t)rt * 16) * kN,
                     (unsigned)(size_t)(void*)apanel, kN, 16, kN);
    asm volatile("s_wait_tensorcnt 0x0" ::: "memory");
#else
    const __bf16* abase = att + ((size_t)c * kN + (size_t)rt * 16 + rloc) * kN + khalf * 8;
#endif
    v16bf bcur = *(const v16bf*)(bptr);
    for (int kt = 0; kt < 64; ++kt) {
#if TDM_OK
      const unsigned raw = rawbase + (unsigned)kt * 64u;
      const v8bf lo = *(const v8bf*)(apanel + lds_pad(raw));
      const v8bf hi = *(const v8bf*)(apanel + lds_pad(raw + 32u));
#else
      const __bf16* ap = abase + kt * 32;
      const v8bf lo = *(const v8bf*)(ap);
      const v8bf hi = *(const v8bf*)(ap + 16);
#endif
      const v16bf a = __builtin_shufflevector(lo, hi, 0, 1, 2, 3, 4, 5, 6, 7,
                                              8, 9, 10, 11, 12, 13, 14, 15);
      const v16bf bnext = *(const v16bf*)(bptr + (size_t)(((kt + 1) & 63)) * 512);
      acc = __builtin_amdgcn_wmma_f32_16x16x32_bf16(false, a, false, bcur,
                                                    (short)0, acc, false, false);
      bcur = bnext;
    }
  }
#pragma unroll
  for (int v = 0; v < 8; ++v)
    hp[(size_t)(rt * 16 + v + khalf * 8) * kD2 + rloc] = acc[v];
}

// -------------------- batch norm stats over 2048 rows --------------------
__global__ void k_bnstats(const float* __restrict__ hp, float* __restrict__ stats) {
  const int col = blockIdx.x, t = threadIdx.x;
  float s = 0.f, s2 = 0.f;
  for (int n = t; n < kN; n += 256) {
    const float v = hp[(size_t)n * kD2 + col];
    s += v; s2 += v * v;
  }
  __shared__ float r0[256], r1[256];
  r0[t] = s; r1[t] = s2;
  __syncthreads();
  for (int st = 128; st > 0; st >>= 1) {
    if (t < st) { r0[t] += r0[t + st]; r1[t] += r1[t + st]; }
    __syncthreads();
  }
  if (t == 0) {
    const float m = r0[0] / kN;
    stats[col] = m;
    stats[16 + col] = r1[0] / kN - m * m;
  }
}

// -------------------- bn + elu + batch-specific scale/shift + relu ---------
__global__ void k_post(const float* __restrict__ hp, const float* __restrict__ stats,
                       const float* __restrict__ g, const float* __restrict__ b,
                       const float* __restrict__ bscale, const float* __restrict__ bshift,
                       const int* __restrict__ depth, float* __restrict__ xp) {
  const int i = blockIdx.x * 256 + threadIdx.x;  // < N*16
  const int n = i >> 4, f = i & 15;
  float x = (hp[i] - stats[f]) * rsqrtf(stats[16 + f] + 1e-5f) * g[f] + b[f];
  x = d_elu(x);
  const int d = depth[n];
  x = x * bscale[d * kD2 + f] + bshift[d * kD2 + f];
  xp[i] = fmaxf(x, 0.f);
}

// -------------------- per-node aggregator MLP + segment sum ----------------
__global__ void k_agg(const float* __restrict__ xp, const float* __restrict__ w1,
                      const float* __restrict__ b1, const float* __restrict__ w2,
                      const float* __restrict__ b2, const int* __restrict__ gidx,
                      float* __restrict__ gs_acc) {
  const int n = blockIdx.x, t = threadIdx.x;  // blockDim = 64
  __shared__ float xs[16], g1[64], g2[64];
  if (t < 16) xs[t] = xp[(size_t)n * kD2 + t];
  __syncthreads();
  float acc = b1[t];
#pragma unroll
  for (int k = 0; k < 16; ++k) acc += xs[k] * w1[k * 64 + t];
  g1[t] = fmaxf(acc, 0.f);
  __syncthreads();
  float acc2 = b2[t];
#pragma unroll 16
  for (int k = 0; k < 64; ++k) acc2 += g1[k] * w2[k * 64 + t];
  g2[t] = acc2;
  __syncthreads();
  if (t < 32) {
    const float ns = g2[32 + t] * (1.f / (1.f + expf(-g2[t])));
    atomicAdd(&gs_acc[gidx[n] * 32 + t], ns);
  }
}

// -------------------- final graph MLP + batch norm --------------------
__global__ void k_final(const float* __restrict__ gs_acc, const float* __restrict__ w1,
                        const float* __restrict__ b1, const float* __restrict__ w2,
                        const float* __restrict__ b2, const float* __restrict__ g,
                        const float* __restrict__ bb, float* __restrict__ out) {
  const int t = threadIdx.x;  // 256 threads
  __shared__ float gs[256], h1[256], h2[128], mv[32];
  gs[t] = gs_acc[t];
  __syncthreads();
  {
    const int i = t >> 5, j = t & 31;
    float acc = b1[j];
#pragma unroll
    for (int k = 0; k < 32; ++k) acc += gs[i * 32 + k] * w1[k * 32 + j];
    h1[t] = fmaxf(acc, 0.f);
  }
  __syncthreads();
  if (t < 128) {
    const int i = t >> 4, j = t & 15;
    float acc = b2[j];
#pragma unroll
    for (int k = 0; k < 32; ++k) acc += h1[i * 32 + k] * w2[k * 16 + j];
    h2[t] = acc;
  }
  __syncthreads();
  if (t < 16) {
    float s = 0.f, s2 = 0.f;
#pragma unroll
    for (int i = 0; i < kNG; ++i) { const float v = h2[i * 16 + t]; s += v; s2 += v * v; }
    const float m = s / kNG;
    mv[t] = m;
    mv[16 + t] = s2 / kNG - m * m;
  }
  __syncthreads();
  if (t < 128) {
    const int j = t & 15;
    out[t] = (h2[t] - mv[j]) * rsqrtf(mv[16 + j] + 1e-5f) * g[j] + bb[j];
  }
}

// ---------------------------------------------------------------------------
extern "C" void kernel_launch(void* const* d_in, const int* in_sizes, int n_in,
                              void* d_out, int out_size, void* d_ws, size_t ws_size,
                              hipStream_t stream) {
  (void)in_sizes; (void)n_in; (void)out_size; (void)ws_size;
  const float* nf     = (const float*)d_in[0];
  const float* ef     = (const float*)d_in[1];
  const float* Wheads = (const float*)d_in[2];
  const float* aheads = (const float*)d_in[3];
  const float* Wout   = (const float*)d_in[4];
  const float* aout   = (const float*)d_in[5];
  const float* bn1g   = (const float*)d_in[6];
  const float* bn1b   = (const float*)d_in[7];
  const float* bss    = (const float*)d_in[8];
  const float* bsh    = (const float*)d_in[9];
  const float* m1w1   = (const float*)d_in[10];
  const float* m1b1   = (const float*)d_in[11];
  const float* m1w2   = (const float*)d_in[12];
  const float* m1b2   = (const float*)d_in[13];
  const float* m2w1   = (const float*)d_in[14];
  const float* m2b1   = (const float*)d_in[15];
  const float* m2w2   = (const float*)d_in[16];
  const float* m2b2   = (const float*)d_in[17];
  const float* bn2g   = (const float*)d_in[18];
  const float* bn2b   = (const float*)d_in[19];
  const int*   gidx   = (const int*)d_in[20];
  const int*   didx   = (const int*)d_in[21];

  char* ws = (char*)d_ws;
  float*  edge   = (float*)(ws + OFF_EDGE);
  __bf16* att    = (__bf16*)(ws + OFF_ATT);
  float*  xcat   = (float*)(ws + OFF_XCAT);
  float*  Wh     = (float*)(ws + OFF_WH);
  __bf16* bswz   = (__bf16*)(ws + OFF_BSWZ);
  float*  Whout  = (float*)(ws + OFF_WHOUT);
  __bf16* bswzo  = (__bf16*)(ws + OFF_BSWZO);
  float*  s1     = (float*)(ws + OFF_S1);
  float*  s2     = (float*)(ws + OFF_S2);
  float*  a_sum  = (float*)(ws + OFF_ASUM);
  float*  b_sum  = (float*)(ws + OFF_BSUM);
  float*  chans  = (float*)(ws + OFF_CHANS);
  float*  hp     = (float*)(ws + OFF_HP);
  float*  bnst   = (float*)(ws + OFF_BN);
  float*  xp     = (float*)(ws + OFF_XP);
  float*  gs_acc = (float*)(ws + OFF_GS);

  float* out_gs   = (float*)d_out;
  float* out_edge = out_gs + kNG * kD2;

  const int nEN2blk = (int)((kE * kNN) / 256);  // 65536

  k_zero_gs<<<1, 256, 0, stream>>>(gs_acc);

  // initial DSN: edge_attr = dsn(edge_features^T reshaped)
  k_init_edges<<<kN, 256, 0, stream>>>(ef, edge, a_sum);
  k_colsum<<<(kE * kN) / 256, 256, 0, stream>>>(edge, b_sum);
  k_scalars<<<kE, 256, 0, stream>>>(a_sum, b_sum, chans);
  k_apply_init<<<nEN2blk, 256, 0, stream>>>(edge, a_sum, b_sum, chans);

  // 4 attention heads (edge_attr threads through as raw e, pre-DSN)
  for (int head = 0; head < 4; ++head) {
    k_gemm_Wh<<<kN / 4, 256, 0, stream>>>(nf, Wheads + (size_t)head * kD0 * kD1, Wh);
    k_swizzleB<<<(kN * kD1) / 256, 256, 0, stream>>>(Wh, bswz, kD1, 4);
    k_s12<<<kN / 256, 256, 0, stream>>>(Wh, aheads + head * 2 * kD1, s1, s2, kD1);
    k_edge_gat<<<kN, 256, 0, stream>>>(edge, edge, s1, s2, a_sum);
    k_colsum<<<(kE * kN) / 256, 256, 0, stream>>>(edge, b_sum);
    k_scalars<<<kE, 256, 0, stream>>>(a_sum, b_sum, chans);
    k_apply_bf16<<<nEN2blk, 256, 0, stream>>>(edge, att, a_sum, b_sum, chans);
    k_gemm_att<<<dim3(kN / 16, kE), 32, 0, stream>>>(att, bswz, xcat, head);
  }

  // output head: e written straight into d_out (it is the returned edge_attr)
  k_gemm_Whout<<<kN / 16, 256, 0, stream>>>(xcat, Wout, Whout);
  k_swizzleB<<<(kN * kD2) / 256, 256, 0, stream>>>(Whout, bswzo, kD2, 1);
  k_s12<<<kN / 256, 256, 0, stream>>>(Whout, aout, s1, s2, kD2);
  k_edge_gat<<<kN, 256, 0, stream>>>(edge, out_edge, s1, s2, a_sum);
  k_colsum<<<(kE * kN) / 256, 256, 0, stream>>>(out_edge, b_sum);
  k_scalars<<<kE, 256, 0, stream>>>(a_sum, b_sum, chans);
  k_apply_bf16<<<nEN2blk, 256, 0, stream>>>(out_edge, att, a_sum, b_sum, chans);
  k_gemm_out<<<kN / 16, 32, 0, stream>>>(att, bswzo, hp);

  // post: bn1 -> elu -> batch-specific -> relu -> aggregator -> final bn2
  k_bnstats<<<kD2, 256, 0, stream>>>(hp, bnst);
  k_post<<<(kN * kD2) / 256, 256, 0, stream>>>(hp, bnst, bn1g, bn1b, bss, bsh, didx, xp);
  k_agg<<<kN, 64, 0, stream>>>(xp, m1w1, m1b1, m1w2, m1b2, gidx, gs_acc);
  k_final<<<1, 256, 0, stream>>>(gs_acc, m2w1, m2b1, m2w2, m2b2, bn2g, bn2b, out_gs);
}
